// MoEPolicy_77945066487968
// MI455X (gfx1250) — matrix operand order repaired and checked by reading, not compiled, not run on hardware
//
#include <hip/hip_runtime.h>

typedef __attribute__((ext_vector_type(16))) _Float16 v16h;
typedef __attribute__((ext_vector_type(8)))  _Float16 v8h;
typedef __attribute__((ext_vector_type(8)))  float    v8f;

#define WAVES 8
#define NFRAG 16
#define XPITCH 20   // floats per row of the relayout tile (16 + 4 pad: 16B-aligned rows, bank-skewed)
#define LN_EPS 1e-5f

union V16 { v16h v; v8h h[2]; };

__device__ __forceinline__ v8f zero8() {
  v8f c;
#pragma unroll
  for (int i = 0; i < 8; ++i) c[i] = 0.f;
  return c;
}

// WMMA with B fragment loaded from LDS (16 halves = 32B -> two ds_load_b128)
__device__ __forceinline__ v8f wmma_k32(v16h a, const _Float16* fp, v8f c) {
  V16 b;
  b.h[0] = *(const v8h*)fp;
  b.h[1] = *(const v8h*)(fp + 8);
  return __builtin_amdgcn_wmma_f32_16x16x32_f16(false, a, false, b.v,
                                                (short)0, c, false, false);
}

__device__ __forceinline__ void pack8(v16h& a, int base, float4 lo, float4 hi) {
  a[base+0] = (_Float16)lo.x; a[base+1] = (_Float16)lo.y;
  a[base+2] = (_Float16)lo.z; a[base+3] = (_Float16)lo.w;
  a[base+4] = (_Float16)hi.x; a[base+5] = (_Float16)hi.y;
  a[base+6] = (_Float16)hi.z; a[base+7] = (_Float16)hi.w;
}

// Relayout (C layout -> A layout via per-wave LDS tile), then bias + LayerNorm +
// SiLU computed IN-LANE: each lane owns row M=n, hidden slice [half*8, half*8+8).
// Only 2 cross-lane ops (xor-16 to merge the two half-row partial stats).
__device__ __forceinline__ void ln_silu_lane(
    v8f c, float* xt, int n, int half,
    const float* bv, const float* gv, const float* bev, float* y)
{
#pragma unroll
  for (int r = 0; r < 8; ++r) xt[(r + 8*half)*XPITCH + n] = c[r];
  asm volatile("" ::: "memory");   // ds is in-order within a wave; keep compiler honest
  const float* xr = xt + n*XPITCH + half*8;
  const float4 x0 = *(const float4*)(xr);
  const float4 x1 = *(const float4*)(xr + 4);
  const float4 b0 = *(const float4*)(bv + half*8);
  const float4 b1 = *(const float4*)(bv + half*8 + 4);
  float x[8];
  x[0]=x0.x+b0.x; x[1]=x0.y+b0.y; x[2]=x0.z+b0.z; x[3]=x0.w+b0.w;
  x[4]=x1.x+b1.x; x[5]=x1.y+b1.y; x[6]=x1.z+b1.z; x[7]=x1.w+b1.w;
  float s = 0.f, s2 = 0.f;
#pragma unroll
  for (int i = 0; i < 8; ++i) { s += x[i]; s2 = fmaf(x[i], x[i], s2); }
  s  += __shfl_xor(s , 16, 32);
  s2 += __shfl_xor(s2, 16, 32);
  const float mu  = s * 0.0625f;
  const float var = s2 * 0.0625f - mu * mu;
  const float inv = rsqrtf(var + LN_EPS);
  const float4 g0 = *(const float4*)(gv + half*8);
  const float4 g1 = *(const float4*)(gv + half*8 + 4);
  const float4 e0 = *(const float4*)(bev + half*8);
  const float4 e1 = *(const float4*)(bev + half*8 + 4);
  const float gg[8] = {g0.x,g0.y,g0.z,g0.w,g1.x,g1.y,g1.z,g1.w};
  const float ee[8] = {e0.x,e0.y,e0.z,e0.w,e1.x,e1.y,e1.z,e1.w};
#pragma unroll
  for (int i = 0; i < 8; ++i) {
    float t = (x[i] - mu) * inv * gg[i] + ee[i];
    y[i] = t / (1.f + __expf(-t));
  }
}

// In-lane values (row M=n, K slice half*8..+7) -> f16 A fragment, K padded to 32.
__device__ __forceinline__ v16h to_frag(const float* y) {
  v16h a;
#pragma unroll
  for (int i = 0; i < 8; ++i) { a[i] = (_Float16)y[i]; a[i+8] = (_Float16)0.f; }
  return a;
}

// Write 8 router-weight-scaled values into half of a K=64 A fragment.
__device__ __forceinline__ void set8(v16h& a, int base, const float* y, float w) {
#pragma unroll
  for (int i = 0; i < 8; ++i) a[base + i] = (_Float16)(y[i] * w);
}

__global__ __launch_bounds__(256, 1) void moe_policy_kernel(
    const float* __restrict__ obs,
    const float* __restrict__ rW1, const float* __restrict__ rb1,
    const float* __restrict__ rg1, const float* __restrict__ rbe1,
    const float* __restrict__ rW2, const float* __restrict__ rb2,
    const float* __restrict__ eW1, const float* __restrict__ eb1,
    const float* __restrict__ eg1, const float* __restrict__ ebe1,
    const float* __restrict__ eW2, const float* __restrict__ eb2,
    const float* __restrict__ eg2, const float* __restrict__ ebe2,
    const float* __restrict__ hW,  const float* __restrict__ hb,
    float* __restrict__ out, int ntiles)
{
  // f16 B fragments: [0,1]=rW1(K0,K32); [2..9]=eW1 e*2+(K0,K32);
  // [10..13]=eW2 (K padded to 32); [14,15]=hW K-stacked over experts (K=64)
  __shared__ __align__(16) _Float16 fragLDS[NFRAG][32][16];
  __shared__ __align__(16) float xtile[WAVES][16][XPITCH];  // per-wave relayout tile (f32)
  __shared__ __align__(16) float svec[576];

  const int tid  = threadIdx.x;
  const int lane = tid & 31;
  const int wid  = tid >> 5;
  const int n    = lane & 15;
  const int half = lane >> 4;

  // ---- stage small f32 vectors into LDS ----
  for (int i = tid; i < 16; i += 256) {
    svec[0 + i]  = rb1[i];  svec[16 + i] = rg1[i];  svec[32 + i] = rbe1[i];
  }
  for (int i = tid; i < 64; i += 256) {
    svec[48  + i] = rW2[i];
    svec[128 + i] = eb1[i];  svec[192 + i] = eg1[i]; svec[256 + i] = ebe1[i];
    svec[320 + i] = eb2[i];  svec[384 + i] = eg2[i]; svec[448 + i] = ebe2[i];
    svec[512 + i] = hb[i];
  }
  if (tid < 4) svec[112 + tid] = rb2[tid];

  // ---- build f16 WMMA B fragments in LDS (B layout: lane=N, elem i<8 -> K = half*8+i, i>=8 -> K = 16+half*8+i-8) ----
  for (int f = wid; f < NFRAG; f += WAVES) {
    _Float16* dst = fragLDS[f][lane];
    if (f >= 14) {
      // K-stacked head weights: global K in [0,64), expert = K>>4, row = K&15
      int kb = (f - 14) * 32;
#pragma unroll
      for (int i = 0; i < 16; ++i) {
        int K = kb + (i < 8 ? half * 8 + i : 16 + half * 8 + (i - 8));
        dst[i] = (_Float16)hW[(K >> 4) * 256 + (K & 15) * 16 + n];
      }
    } else {
      const float* W; int kb = 0; bool pad = false;
      if (f < 2)       { W = rW1;                         kb = f * 32;             }
      else if (f < 10) { W = eW1 + ((f - 2) >> 1) * 1024; kb = ((f - 2) & 1) * 32; }
      else             { W = eW2 + (f - 10) * 256;        pad = true;              }
#pragma unroll
      for (int i = 0; i < 16; ++i) {
        int K = kb + (i < 8 ? half * 8 + i : 16 + half * 8 + (i - 8));
        float v = (pad && K >= 16) ? 0.f : W[K * 16 + n];
        dst[i] = (_Float16)v;
      }
    }
  }
  __syncthreads();

  float* xt = &xtile[wid][0][0];
  const float4 rb2v = *(const float4*)&svec[112];
  // hoist head-bias slices (per-lane action slice, per expert)
  float4 hbA[4], hbB[4];
#pragma unroll
  for (int e = 0; e < 4; ++e) {
    hbA[e] = *(const float4*)&svec[512 + e * 16 + half * 8];
    hbB[e] = *(const float4*)&svec[512 + e * 16 + half * 8 + 4];
  }

  const int gw     = blockIdx.x * WAVES + wid;
  const int stride = gridDim.x * WAVES;

  for (int tile = gw; tile < ntiles; tile += stride) {
    // ---- obs -> A fragments (lane M=n; elems 0..7 -> K=half*8+i, 8..15 -> K=16+half*8+i) ----
    const float* pr = obs + (size_t)(tile * 16 + n) * 64 + half * 8;
    float4 q0 = *(const float4*)(pr +  0);
    float4 q1 = *(const float4*)(pr +  4);
    float4 q2 = *(const float4*)(pr + 16);
    float4 q3 = *(const float4*)(pr + 20);
    float4 q4 = *(const float4*)(pr + 32);
    float4 q5 = *(const float4*)(pr + 36);
    float4 q6 = *(const float4*)(pr + 48);
    float4 q7 = *(const float4*)(pr + 52);
    v16h a0, a1;
    pack8(a0, 0, q0, q1); pack8(a0, 8, q2, q3);
    pack8(a1, 0, q4, q5); pack8(a1, 8, q6, q7);

    // ---- router backbone ----
    v8f rc = zero8();
    rc = wmma_k32(a0, fragLDS[0][lane], rc);
    rc = wmma_k32(a1, fragLDS[1][lane], rc);
    float ry[8];
    ln_silu_lane(rc, xt, n, half, svec + 0, svec + 16, svec + 32, ry);

    // ---- router head: each lane owns row m=n (half-rows merged by xor-16) ----
    float l0 = 0.f, l1 = 0.f, l2 = 0.f, l3 = 0.f;
#pragma unroll
    for (int i = 0; i < 8; ++i) {
      const float4 w2 = *(const float4*)(svec + 48 + (half * 8 + i) * 4);
      l0 = fmaf(ry[i], w2.x, l0);
      l1 = fmaf(ry[i], w2.y, l1);
      l2 = fmaf(ry[i], w2.z, l2);
      l3 = fmaf(ry[i], w2.w, l3);
    }
    l0 += __shfl_xor(l0, 16, 32); l1 += __shfl_xor(l1, 16, 32);
    l2 += __shfl_xor(l2, 16, 32); l3 += __shfl_xor(l3, 16, 32);
    l0 += rb2v.x; l1 += rb2v.y; l2 += rb2v.z; l3 += rb2v.w;
    // top-2 mask + softmax (once per lane; weights stay in-lane)
    float w0, w1, w2s, w3s;
    {
      float am = fmaxf(l0, l1), bm = fminf(l0, l1);
      float cm = fmaxf(l2, l3), dm = fminf(l2, l3);
      float m1 = fmaxf(am, cm);
      float m2 = fmaxf(fminf(am, cm), fmaxf(bm, dm));   // 2nd largest of 4
      float z0 = (l0 >= m2) ? __expf(l0 - m1) : 0.f;
      float z1 = (l1 >= m2) ? __expf(l1 - m1) : 0.f;
      float z2 = (l2 >= m2) ? __expf(l2 - m1) : 0.f;
      float z3 = (l3 >= m2) ? __expf(l3 - m1) : 0.f;
      float zi = 1.f / (z0 + z1 + z2 + z3);
      w0 = z0 * zi; w1 = z1 * zi; w2s = z2 * zi; w3s = z3 * zi;
    }

    // ---- experts: L1 -> LN -> L2 -> LN -> (w-scaled into K=64 head A fragment) ----
    v16h a30, a31;
#pragma unroll
    for (int e = 0; e < 4; ++e) {
      v8f c1 = zero8();
      c1 = wmma_k32(a0, fragLDS[2 + 2 * e][lane], c1);
      c1 = wmma_k32(a1, fragLDS[3 + 2 * e][lane], c1);
      float y[8];
      ln_silu_lane(c1, xt, n, half,
                   svec + 128 + e * 16, svec + 192 + e * 16, svec + 256 + e * 16, y);
      v16h a2 = to_frag(y);
      v8f c2 = zero8();
      c2 = wmma_k32(a2, fragLDS[10 + e][lane], c2);
      ln_silu_lane(c2, xt, n, half,
                   svec + 320 + e * 16, svec + 384 + e * 16, svec + 448 + e * 16, y);
      const float we = (e == 0) ? w0 : (e == 1) ? w1 : (e == 2) ? w2s : w3s;
      if (e == 0) set8(a30, 0, y, we);
      else if (e == 1) set8(a30, 8, y, we);
      else if (e == 2) set8(a31, 0, y, we);
      else set8(a31, 8, y, we);
    }

    // ---- fused head: out = [w0*h2_0 | w1*h2_1 | w2*h2_2 | w3*h2_3] @ stacked(hW) ----
    v8f c3 = zero8();
    c3 = wmma_k32(a30, fragLDS[14][lane], c3);
    c3 = wmma_k32(a31, fragLDS[15][lane], c3);

    // relayout to row-per-lane, add weighted head bias, coalesced b128 stores
#pragma unroll
    for (int r = 0; r < 8; ++r) xt[(r + 8 * half) * XPITCH + n] = c3[r];
    asm volatile("" ::: "memory");
    const float* orow = xt + n * XPITCH + half * 8;
    float4 o0 = *(const float4*)(orow);
    float4 o1 = *(const float4*)(orow + 4);
#pragma unroll
    for (int e = 0; e < 4; ++e) {
      const float we = (e == 0) ? w0 : (e == 1) ? w1 : (e == 2) ? w2s : w3s;
      o0.x = fmaf(we, hbA[e].x, o0.x); o0.y = fmaf(we, hbA[e].y, o0.y);
      o0.z = fmaf(we, hbA[e].z, o0.z); o0.w = fmaf(we, hbA[e].w, o0.w);
      o1.x = fmaf(we, hbB[e].x, o1.x); o1.y = fmaf(we, hbB[e].y, o1.y);
      o1.z = fmaf(we, hbB[e].z, o1.z); o1.w = fmaf(we, hbB[e].w, o1.w);
    }
    float* po = out + (size_t)tile * 256 + n * 16 + half * 8;
    *(float4*)(po)     = o0;
    *(float4*)(po + 4) = o1;
  }
}

extern "C" void kernel_launch(void* const* d_in, const int* in_sizes, int n_in,
                              void* d_out, int out_size, void* d_ws, size_t ws_size,
                              hipStream_t stream) {
  const float* obs  = (const float*)d_in[0];
  const float* rW1  = (const float*)d_in[1];
  const float* rb1  = (const float*)d_in[2];
  const float* rg1  = (const float*)d_in[3];
  const float* rbe1 = (const float*)d_in[4];
  const float* rW2  = (const float*)d_in[5];
  const float* rb2  = (const float*)d_in[6];
  const float* eW1  = (const float*)d_in[7];
  const float* eb1  = (const float*)d_in[8];
  const float* eg1  = (const float*)d_in[9];
  const float* ebe1 = (const float*)d_in[10];
  const float* eW2  = (const float*)d_in[11];
  const float* eb2  = (const float*)d_in[12];
  const float* eg2  = (const float*)d_in[13];
  const float* ebe2 = (const float*)d_in[14];
  const float* hW   = (const float*)d_in[15];
  const float* hb   = (const float*)d_in[16];

  const int Bsz    = in_sizes[0] / 64;
  const int ntiles = Bsz / 16;
  int blocks = (ntiles + WAVES - 1) / WAVES;
  if (blocks > 4096) blocks = 4096;

  hipLaunchKernelGGL(moe_policy_kernel, dim3(blocks), dim3(256), 0, stream,
                     obs, rW1, rb1, rg1, rbe1, rW2, rb2,
                     eW1, eb1, eg1, ebe1, eW2, eb2, eg2, ebe2, hW, hb,
                     (float*)d_out, ntiles);
}
